// CrossAttention_50577534878250
// MI455X (gfx1250) — compile-verified, matrix operand order checked by reading, not run on hardware
//
#include <hip/hip_runtime.h>
#include <hip/hip_bf16.h>

// ---------------------------------------------------------------------------
// CDNA5 (gfx1250) fused cross-attention, f16 WMMA everywhere,
// async global->LDS tile staging where available.
// ---------------------------------------------------------------------------

#define B_ 16
#define C_ 512
#define HW_ 4096
#define HEADS_ 8
#define HD_ 64
#define NA_ 64           // anchors per batch (8x8 pool)
#define SCALE_ 0.125f    // 64^-0.5

typedef __attribute__((ext_vector_type(8)))  _Float16 v8h;
typedef __attribute__((ext_vector_type(16))) _Float16 v16h;
typedef __attribute__((ext_vector_type(8)))  float    v8f;
typedef __attribute__((ext_vector_type(4)))  int      v4i;

// --- async global->LDS (GLOBAL_LOAD_ASYNC_TO_LDS_B128, ASYNCcnt-tracked) ---
#if defined(__gfx1250__) && __has_builtin(__builtin_amdgcn_global_load_async_to_lds_b128)
#define USE_ASYNC_LDS 1
#else
#define USE_ASYNC_LDS 0
#endif

#if USE_ASYNC_LDS
typedef __attribute__((address_space(1))) v4i gv4i_t;   // global int4
typedef __attribute__((address_space(3))) v4i lv4i_t;   // LDS int4
#endif

// Copy 16 bytes global -> LDS. Generic LDS pointers carry the LDS byte offset
// in bits[31:0] (flat aperture rules), so inttoptr to AS(3) is the offset ptr.
__device__ __forceinline__ void cp16_async(const _Float16* g, _Float16* l) {
#if USE_ASYNC_LDS
    __builtin_amdgcn_global_load_async_to_lds_b128(
        (gv4i_t*)(uintptr_t)g,
        (lv4i_t*)(uint32_t)(uintptr_t)l,
        0, 0);
#else
    *(v8h*)l = *(const v8h*)g;
#endif
}

__device__ __forceinline__ void cp_async_wait() {
#if USE_ASYNC_LDS
#if __has_builtin(__builtin_amdgcn_s_wait_asynccnt)
    __builtin_amdgcn_s_wait_asynccnt(0);
#else
    asm volatile("s_wait_asynccnt 0" ::: "memory");
#endif
#endif
}

// Build a 16x32 A-fragment (or 32x16 B-fragment) from LDS, per the CDNA5 ISA
// 16-bit A layout: lane holds row (M or N) = lane&15; this lane group's 8
// K-halves start at klo, the second 8 at klo+16.  Two ds_load_b128.
__device__ __forceinline__ v16h ld_frag(const _Float16* p) {
    union { v16h v; v8h h[2]; } u;
    u.h[0] = *(const v8h*)p;
    u.h[1] = *(const v8h*)(p + 16);
    return u.v;
}

__device__ __forceinline__ v8f wmma16(v16h a, v16h b, v8f c) {
    return __builtin_amdgcn_wmma_f32_16x16x32_f16(
        /*neg_a=*/false, a, /*neg_b=*/false, b,
        /*c_mod=*/(short)0, c, /*reuse_a=*/false, /*reuse_b=*/false);
}

// ---------------------------------------------------------------------------
__global__ void __launch_bounds__(256) cvt_f16_kernel(
    const float* __restrict__ src, _Float16* __restrict__ dst, int n) {
    int i = blockIdx.x * 256 + threadIdx.x;
    if (i < n) dst[i] = (_Float16)src[i];
}

// feat [B,C,64,64] -> pooled [B,C,64] (anchor a = py*8+px), blocked mean.
__global__ void __launch_bounds__(256) pool_kernel(
    const float* __restrict__ feat, float* __restrict__ pooled) {
    int i = blockIdx.x * 256 + threadIdx.x;       // over B*C*64
    if (i >= B_ * C_ * NA_) return;
    int a = i & 63, bc = i >> 6;
    int py = a >> 3, px = a & 7;
    const float* base = feat + (size_t)bc * HW_ + py * 8 * 64 + px * 8;
    float s = 0.f;
#pragma unroll
    for (int iy = 0; iy < 8; ++iy)
#pragma unroll
        for (int ix = 0; ix < 8; ++ix) s += base[iy * 64 + ix];
    pooled[i] = s * (1.0f / 64.0f);
}

// ---------------------------------------------------------------------------
// Fused LayerNorm + GEMM:  Y[b, tok, n] = LN(X[b, :, tok]) @ W[n, :]^T + bias
// X layout [B, 512, HWLEN] f32 ; W f16 [NOUT, 512] ; Y f16 [B*HWLEN, NOUT].
// Block: 256 threads, tile 128 tokens x 128 cols, K=512 fully resident in LDS.
// HWLEN/NOUT are template constants so store addressing is shift+add and the
// bounds guard folds away for full tiles.
// Dynamic LDS: A 128KB + B 128KB + gamma/beta/stats  (= 269312 bytes)
// ---------------------------------------------------------------------------
template <int HWLEN, int NOUT>
__global__ void __launch_bounds__(256) ln_gemm_kernel(
    const float* __restrict__ X,
    const float* __restrict__ gamma, const float* __restrict__ beta,
    const _Float16* __restrict__ Wt, const float* __restrict__ bias,
    _Float16* __restrict__ Y) {
    extern __shared__ char smem[];
    _Float16* Alds = (_Float16*)smem;                  // [128][512]
    _Float16* Blds = (_Float16*)(smem + 131072);       // [128][512]
    float* gs    = (float*)(smem + 262144);            // [512]
    float* bs    = (float*)(smem + 264192);            // [512]
    float* part  = (float*)(smem + 266240);            // [256][2]
    float* meanv = (float*)(smem + 268288);            // [128]
    float* rstdv = (float*)(smem + 268800);            // [128]

    const int tid  = threadIdx.x;
    const int b    = blockIdx.z;
    const int tok0 = blockIdx.x * 128;
    const int n0   = blockIdx.y * 128;

    for (int i = tid; i < C_; i += 256) { gs[i] = gamma[i]; bs[i] = beta[i]; }

    // Weight tile -> LDS via async DMA (no VGPR bounce).
    {
        const _Float16* src = Wt + (size_t)n0 * C_;
        for (int i = tid; i < 128 * C_ / 8; i += 256)
            cp16_async(src + (size_t)i * 8, Blds + i * 8);
    }

    // Coalesced load of A tile + per-token partial LN stats.
    // token n = tid&127 is constant per thread; channels c = (tid>>7) + 2*i.
    {
        const int n = tid & 127;
        const int tok = tok0 + n;
        const bool valid = ((HWLEN & 127) == 0) || (tok < HWLEN);
        const float* xb = X + (size_t)b * C_ * HWLEN + tok;
        float s = 0.f, s2 = 0.f;
        for (int i = 0; i < C_ / 2; ++i) {
            int c = (tid >> 7) + 2 * i;
            float v = valid ? xb[(size_t)c * HWLEN] : 0.f;
            s += v; s2 += v * v;
            Alds[n * C_ + c] = (_Float16)v;
        }
        part[tid * 2] = s; part[tid * 2 + 1] = s2;
    }
    cp_async_wait();
    __syncthreads();
    if (tid < 128) {
        float s  = part[tid * 2]     + part[(tid + 128) * 2];
        float s2 = part[tid * 2 + 1] + part[(tid + 128) * 2 + 1];
        float m  = s * (1.f / C_);
        float var = s2 * (1.f / C_) - m * m;
        meanv[tid] = m;
        rstdv[tid] = rsqrtf(var + 1e-5f);
    }
    __syncthreads();
    // Apply LN in place (each thread owns its own elements).
    {
        const int n = tid & 127;
        const float m = meanv[n], r = rstdv[n];
        for (int i = 0; i < C_ / 2; ++i) {
            int c = (tid >> 7) + 2 * i;
            float v = (float)Alds[n * C_ + c];
            Alds[n * C_ + c] = (_Float16)((v - m) * r * gs[c] + bs[c]);
        }
    }
    __syncthreads();

    // WMMA: 8 waves, each 32(M) x 64(N) = 2x4 tiles of 16x16.
    const int w = tid >> 5, lane = tid & 31;
    const int wm = (w & 3) * 32, wn = (w >> 2) * 64;
    const int row_lo = lane & 15;
    const int klo    = (lane >> 4) * 8;
    const int m_lane = (lane >> 4) * 8;
    const int n_lane = lane & 15;

    v8f acc[2][4] = {};
    for (int k = 0; k < C_; k += 32) {
        v16h a[2], bb[4];
#pragma unroll
        for (int mi = 0; mi < 2; ++mi)
            a[mi] = ld_frag(&Alds[(wm + mi * 16 + row_lo) * C_ + k + klo]);
#pragma unroll
        for (int ni = 0; ni < 4; ++ni)
            bb[ni] = ld_frag(&Blds[(wn + ni * 16 + row_lo) * C_ + k + klo]);
#pragma unroll
        for (int mi = 0; mi < 2; ++mi)
#pragma unroll
            for (int ni = 0; ni < 4; ++ni)
                acc[mi][ni] = wmma16(a[mi], bb[ni], acc[mi][ni]);
    }
#pragma unroll
    for (int mi = 0; mi < 2; ++mi)
#pragma unroll
        for (int ni = 0; ni < 4; ++ni) {
            const int ncol = n0 + wn + ni * 16 + n_lane;
            const float bv = bias[ncol];
            const int trow = tok0 + wm + mi * 16 + m_lane;
            _Float16* yp = Y + ((size_t)b * HWLEN + trow) * NOUT + ncol;
#pragma unroll
            for (int r = 0; r < 8; ++r) {
                if (((HWLEN & 127) == 0) || (trow + r < HWLEN))
                    yp[(size_t)r * NOUT] = (_Float16)(acc[mi][ni][r] + bv);
            }
        }
}

// ---------------------------------------------------------------------------
// Attention 1: per (b,h): out1 = softmax(scale * Ad_h @ K_h^T) @ V_h
// 64 anchors x 4096 keys, online softmax over 32 tiles of 128 keys.
// Block = 128 threads (4 waves), wave owns 16 anchors.
// out1 stored TRANSPOSED [bh][hd][anchor] for attn2's B-fragments.
// ---------------------------------------------------------------------------
__global__ void __launch_bounds__(128) attn1_kernel(
    const _Float16* __restrict__ Ad,    // [B*64, 512]
    const _Float16* __restrict__ Ykv,   // [B*4096, 1024]
    _Float16* __restrict__ out1T) {     // [B*8][64 hd][64 anchor]
    __shared__ _Float16 Ad_s[64 * 64];      // [anchor][hd], pre-scaled
    __shared__ _Float16 K_s[128 * 64];      // [key][hd]
    __shared__ _Float16 Vt_s[64 * 128];     // [hd][key]
    __shared__ _Float16 P_s[64 * 128];      // [anchor][key]

    const int bh = blockIdx.x;
    const int b = bh >> 3, h = bh & 7;
    const int tid = threadIdx.x;
    const int wa = tid >> 5, lane = tid & 31;
    const int row_lo = lane & 15;
    const int klo    = (lane >> 4) * 8;
    const int m_lane = (lane >> 4) * 8;
    const int n_lane = lane & 15;

    for (int i = tid; i < 64 * 64; i += 128) {
        int a = i >> 6, d = i & 63;
        Ad_s[i] = (_Float16)((float)Ad[((size_t)(b * 64 + a)) * C_ + h * 64 + d] * SCALE_);
    }

    float m_run[8], l_run[8];
#pragma unroll
    for (int r = 0; r < 8; ++r) { m_run[r] = -1e30f; l_run[r] = 0.f; }
    v8f oacc[4] = {};

    for (int nt = 0; nt < 32; ++nt) {
        __syncthreads();
        const size_t kvb = ((size_t)b * HW_ + nt * 128) * 1024 + h * 64;
        // K tile via async DMA (16B chunks: key = i>>3, 8-half chunk (i&7)*8)
        for (int i = tid; i < 128 * 8; i += 128) {
            int key = i >> 3, d8 = (i & 7) * 8;
            cp16_async(Ykv + kvb + (size_t)key * 1024 + d8, K_s + key * 64 + d8);
        }
        // V tile, transposed into LDS (element gather, cannot DMA)
        for (int i = tid; i < 128 * 64; i += 128) {
            int key = i >> 6, d = i & 63;
            Vt_s[d * 128 + key] = Ykv[kvb + 512 + (size_t)key * 1024 + d];
        }
        // prefetch next tile (global_prefetch_b8): 128 threads x 128B = 16KB each
        if (nt < 31) {
            __builtin_prefetch((const void*)(Ykv + kvb + 128 * 1024 + tid * 64), 0, 1);
            __builtin_prefetch((const void*)(Ykv + kvb + 128 * 1024 + 512 + tid * 64), 0, 1);
        }
        cp_async_wait();
        __syncthreads();

        // S = Ad @ K^T : wave -> 16 anchors x 128 keys (8 N-tiles)
        v8f s[8];
#pragma unroll
        for (int ni = 0; ni < 8; ++ni) {
            v8f acc = {};
#pragma unroll
            for (int k = 0; k < 64; k += 32) {
                v16h a  = ld_frag(&Ad_s[(wa * 16 + row_lo) * 64 + k + klo]);
                v16h bb = ld_frag(&K_s[(ni * 16 + row_lo) * 64 + k + klo]);
                acc = wmma16(a, bb, acc);
            }
            s[ni] = acc;
        }
        // online softmax (row = m_lane + r, 16-lane shuffle reductions)
#pragma unroll
        for (int r = 0; r < 8; ++r) {
            float mx = m_run[r];
#pragma unroll
            for (int ni = 0; ni < 8; ++ni) mx = fmaxf(mx, s[ni][r]);
            for (int msk = 1; msk < 16; msk <<= 1)
                mx = fmaxf(mx, __shfl_xor(mx, msk, 32));
            float corr = __expf(m_run[r] - mx);
            float sum = 0.f;
#pragma unroll
            for (int ni = 0; ni < 8; ++ni) {
                float p = __expf(s[ni][r] - mx);
                s[ni][r] = p; sum += p;
            }
            for (int msk = 1; msk < 16; msk <<= 1)
                sum += __shfl_xor(sum, msk, 32);
            l_run[r] = l_run[r] * corr + sum;
            m_run[r] = mx;
#pragma unroll
            for (int oi = 0; oi < 4; ++oi) oacc[oi][r] *= corr;
        }
        // P -> LDS (wave-private rows), then oacc += P @ V
#pragma unroll
        for (int ni = 0; ni < 8; ++ni)
#pragma unroll
            for (int r = 0; r < 8; ++r)
                P_s[(wa * 16 + m_lane + r) * 128 + ni * 16 + n_lane] =
                    (_Float16)s[ni][r];
#pragma unroll
        for (int oi = 0; oi < 4; ++oi) {
            v8f acc = oacc[oi];
#pragma unroll
            for (int k = 0; k < 128; k += 32) {
                v16h a  = ld_frag(&P_s[(wa * 16 + row_lo) * 128 + k + klo]);
                v16h bb = ld_frag(&Vt_s[(oi * 16 + row_lo) * 128 + k + klo]);
                acc = wmma16(a, bb, acc);
            }
            oacc[oi] = acc;
        }
    }
#pragma unroll
    for (int oi = 0; oi < 4; ++oi)
#pragma unroll
        for (int r = 0; r < 8; ++r) {
            int hd = oi * 16 + n_lane;
            int a  = wa * 16 + m_lane + r;
            out1T[((size_t)bh * 64 + hd) * 64 + a] =
                (_Float16)(oacc[oi][r] / l_run[r]);
        }
}

// ---------------------------------------------------------------------------
// Attention 2: out2 = softmax(scale * Q @ Ac^T) @ out1   (softmax over 64)
// Block = 256 threads, one (b,h) x 128-token tile; wave owns 16 tokens.
// ---------------------------------------------------------------------------
__global__ void __launch_bounds__(256) attn2_kernel(
    const _Float16* __restrict__ Yq,     // [B*4096, 512]
    const _Float16* __restrict__ Ac,     // [B*64, 512]
    const _Float16* __restrict__ out1T,  // [B*8][64 hd][64 anchor]
    _Float16* __restrict__ Yout2) {      // [B*4096, 512]
    __shared__ _Float16 Ac_s[64 * 64];   // [anchor][hd], pre-scaled
    __shared__ _Float16 O1_s[64 * 64];   // [hd][anchor]
    __shared__ _Float16 Q_s[128 * 64];   // [token][hd]
    __shared__ _Float16 P_s[128 * 64];   // [token][anchor]

    const int tt = blockIdx.x, bh = blockIdx.y;
    const int b = bh >> 3, h = bh & 7;
    const int tid = threadIdx.x;
    const int wa = tid >> 5, lane = tid & 31;
    const int row_lo = lane & 15;
    const int klo    = (lane >> 4) * 8;
    const int m_lane = (lane >> 4) * 8;
    const int n_lane = lane & 15;
    const int tok0 = tt * 128;

    // out1T copy via async DMA (linear 8KB)
    for (int i = tid; i < 64 * 64 / 8; i += 256)
        cp16_async(out1T + (size_t)bh * 4096 + i * 8, O1_s + i * 8);
    // Q tile via async DMA (16B chunks: token = i>>3)
    for (int i = tid; i < 128 * 8; i += 256) {
        int t = i >> 3, d8 = (i & 7) * 8;
        cp16_async(Yq + ((size_t)(b * HW_ + tok0 + t)) * C_ + h * 64 + d8,
                   Q_s + t * 64 + d8);
    }
    // anchors (scaled) - element-wise
    for (int i = tid; i < 64 * 64; i += 256) {
        int a = i >> 6, d = i & 63;
        Ac_s[i] = (_Float16)((float)Ac[((size_t)(b * 64 + a)) * C_ + h * 64 + d] * SCALE_);
    }
    cp_async_wait();
    __syncthreads();

    v8f s[4];
#pragma unroll
    for (int ni = 0; ni < 4; ++ni) {
        v8f acc = {};
#pragma unroll
        for (int k = 0; k < 64; k += 32) {
            v16h a  = ld_frag(&Q_s[(wa * 16 + row_lo) * 64 + k + klo]);
            v16h bb = ld_frag(&Ac_s[(ni * 16 + row_lo) * 64 + k + klo]);
            acc = wmma16(a, bb, acc);
        }
        s[ni] = acc;
    }
    float inv_l[8];
#pragma unroll
    for (int r = 0; r < 8; ++r) {
        float mx = -1e30f;
#pragma unroll
        for (int ni = 0; ni < 4; ++ni) mx = fmaxf(mx, s[ni][r]);
        for (int msk = 1; msk < 16; msk <<= 1)
            mx = fmaxf(mx, __shfl_xor(mx, msk, 32));
        float sum = 0.f;
#pragma unroll
        for (int ni = 0; ni < 4; ++ni) {
            float p = __expf(s[ni][r] - mx);
            s[ni][r] = p; sum += p;
        }
        for (int msk = 1; msk < 16; msk <<= 1)
            sum += __shfl_xor(sum, msk, 32);
        inv_l[r] = 1.f / sum;
    }
#pragma unroll
    for (int ni = 0; ni < 4; ++ni)
#pragma unroll
        for (int r = 0; r < 8; ++r)
            P_s[(wa * 16 + m_lane + r) * 64 + ni * 16 + n_lane] = (_Float16)s[ni][r];

#pragma unroll
    for (int oi = 0; oi < 4; ++oi) {
        v8f acc = {};
#pragma unroll
        for (int k = 0; k < 64; k += 32) {
            v16h a  = ld_frag(&P_s[(wa * 16 + row_lo) * 64 + k + klo]);
            v16h bb = ld_frag(&O1_s[(oi * 16 + row_lo) * 64 + k + klo]);
            acc = wmma16(a, bb, acc);
        }
#pragma unroll
        for (int r = 0; r < 8; ++r) {
            int hd  = oi * 16 + n_lane;
            int tok = tok0 + wa * 16 + m_lane + r;
            Yout2[((size_t)(b * HW_) + tok) * C_ + h * 64 + hd] =
                (_Float16)(acc[r] * inv_l[r]);
        }
    }
}

// ---------------------------------------------------------------------------
// Final: out[b,c,tok] = feat_diff[b,c,tok] + (Yout2 @ Wo^T + bo)[b,tok,c]
// Same tiling as ln_gemm, no LN; stores transposed + residual directly
// (per-lane stores hit 8 consecutive tokens).  Dynamic LDS 262144 bytes.
// ---------------------------------------------------------------------------
__global__ void __launch_bounds__(256) final_gemm_kernel(
    const _Float16* __restrict__ Yout2, const _Float16* __restrict__ Wo,
    const float* __restrict__ bo, const float* __restrict__ feat_diff,
    float* __restrict__ out) {
    extern __shared__ char smem[];
    _Float16* Alds = (_Float16*)smem;              // [128][512]
    _Float16* Blds = (_Float16*)(smem + 131072);   // [128][512]
    const int b = blockIdx.z;
    const int tok0 = blockIdx.x * 128;
    const int n0   = blockIdx.y * 128;
    const int tid = threadIdx.x;

    {
        const _Float16* srcA = Yout2 + ((size_t)(b * HW_) + tok0) * C_;
        const _Float16* srcB = Wo + (size_t)n0 * C_;
        for (int i = tid; i < 128 * C_ / 8; i += 256) {
            cp16_async(srcA + (size_t)i * 8, Alds + i * 8);
            cp16_async(srcB + (size_t)i * 8, Blds + i * 8);
        }
    }
    cp_async_wait();
    __syncthreads();

    const int w = tid >> 5, lane = tid & 31;
    const int wm = (w & 3) * 32, wn = (w >> 2) * 64;
    const int row_lo = lane & 15;
    const int klo    = (lane >> 4) * 8;
    const int m_lane = (lane >> 4) * 8;
    const int n_lane = lane & 15;

    v8f acc[2][4] = {};
    for (int k = 0; k < C_; k += 32) {
        v16h a[2], bb[4];
#pragma unroll
        for (int mi = 0; mi < 2; ++mi)
            a[mi] = ld_frag(&Alds[(wm + mi * 16 + row_lo) * C_ + k + klo]);
#pragma unroll
        for (int ni = 0; ni < 4; ++ni)
            bb[ni] = ld_frag(&Blds[(wn + ni * 16 + row_lo) * C_ + k + klo]);
#pragma unroll
        for (int mi = 0; mi < 2; ++mi)
#pragma unroll
            for (int ni = 0; ni < 4; ++ni)
                acc[mi][ni] = wmma16(a[mi], bb[ni], acc[mi][ni]);
    }
#pragma unroll
    for (int mi = 0; mi < 2; ++mi)
#pragma unroll
        for (int ni = 0; ni < 4; ++ni) {
            const int c = n0 + wn + ni * 16 + n_lane;
            const float bv = bo[c];
            const int trow = tok0 + wm + mi * 16 + m_lane;
            const size_t base = ((size_t)(b * C_) + c) * HW_ + trow;
#pragma unroll
            for (int r = 0; r < 8; ++r)
                out[base + r] = feat_diff[base + r] + acc[mi][ni][r] + bv;
        }
}

// ---------------------------------------------------------------------------
extern "C" void kernel_launch(void* const* d_in, const int* in_sizes, int n_in,
                              void* d_out, int out_size, void* d_ws, size_t ws_size,
                              hipStream_t stream) {
    (void)in_sizes; (void)n_in; (void)out_size; (void)ws_size;
    const float* feat_diff = (const float*)d_in[0];
    const float* feat_con  = (const float*)d_in[1];
    const float* g1 = (const float*)d_in[2];  const float* b1 = (const float*)d_in[3];
    const float* w_diff = (const float*)d_in[4]; const float* bd = (const float*)d_in[5];
    const float* g2 = (const float*)d_in[6];  const float* b2 = (const float*)d_in[7];
    const float* wq = (const float*)d_in[8];  const float* bq = (const float*)d_in[9];
    const float* g3 = (const float*)d_in[10]; const float* b3 = (const float*)d_in[11];
    const float* w_con = (const float*)d_in[12]; const float* bc = (const float*)d_in[13];
    const float* g4 = (const float*)d_in[14]; const float* b4 = (const float*)d_in[15];
    const float* w_kv = (const float*)d_in[16]; const float* bkv = (const float*)d_in[17];
    const float* w_o = (const float*)d_in[18]; const float* bo = (const float*)d_in[19];
    float* out = (float*)d_out;

    // workspace layout
    char* ws = (char*)d_ws;
    size_t off = 0;
    _Float16* wq_h   = (_Float16*)(ws + off); off += (size_t)512 * 512 * 2;
    _Float16* wkv_h  = (_Float16*)(ws + off); off += (size_t)1024 * 512 * 2;
    _Float16* wdiff_h= (_Float16*)(ws + off); off += (size_t)512 * 512 * 2;
    _Float16* wcon_h = (_Float16*)(ws + off); off += (size_t)512 * 512 * 2;
    _Float16* wo_h   = (_Float16*)(ws + off); off += (size_t)512 * 512 * 2;
    float* pooled_d  = (float*)(ws + off);    off += (size_t)B_ * C_ * NA_ * 4;
    float* pooled_c  = (float*)(ws + off);    off += (size_t)B_ * C_ * NA_ * 4;
    _Float16* Ad     = (_Float16*)(ws + off); off += (size_t)B_ * NA_ * C_ * 2;
    _Float16* Ac     = (_Float16*)(ws + off); off += (size_t)B_ * NA_ * C_ * 2;
    _Float16* out1T  = (_Float16*)(ws + off); off += (size_t)B_ * HEADS_ * 64 * 64 * 2;
    _Float16* Yq     = (_Float16*)(ws + off); off += (size_t)B_ * HW_ * C_ * 2;
    _Float16* Yout2  = (_Float16*)(ws + off); off += (size_t)B_ * HW_ * C_ * 2;
    _Float16* Ykv    = (_Float16*)(ws + off); off += (size_t)B_ * HW_ * 1024 * 2;

    const size_t LN_SMEM = 269312;
    const size_t FG_SMEM = 262144;

    // 1) weights -> f16
    cvt_f16_kernel<<<(512*512 + 255)/256, 256, 0, stream>>>(wq, wq_h, 512*512);
    cvt_f16_kernel<<<(1024*512 + 255)/256, 256, 0, stream>>>(w_kv, wkv_h, 1024*512);
    cvt_f16_kernel<<<(512*512 + 255)/256, 256, 0, stream>>>(w_diff, wdiff_h, 512*512);
    cvt_f16_kernel<<<(512*512 + 255)/256, 256, 0, stream>>>(w_con, wcon_h, 512*512);
    cvt_f16_kernel<<<(512*512 + 255)/256, 256, 0, stream>>>(w_o, wo_h, 512*512);

    // 2) pooling (both features)
    pool_kernel<<<(B_*C_*NA_ + 255)/256, 256, 0, stream>>>(feat_diff, pooled_d);
    pool_kernel<<<(B_*C_*NA_ + 255)/256, 256, 0, stream>>>(feat_con, pooled_c);

    // 3) fused LN + projections
    ln_gemm_kernel<NA_, 512><<<dim3(1, 4, B_), 256, LN_SMEM, stream>>>(
        pooled_d, g1, b1, wdiff_h, bd, Ad);
    ln_gemm_kernel<NA_, 512><<<dim3(1, 4, B_), 256, LN_SMEM, stream>>>(
        pooled_c, g3, b3, wcon_h, bc, Ac);
    ln_gemm_kernel<HW_, 512><<<dim3(32, 4, B_), 256, LN_SMEM, stream>>>(
        feat_diff, g2, b2, wq_h, bq, Yq);
    ln_gemm_kernel<HW_, 1024><<<dim3(32, 8, B_), 256, LN_SMEM, stream>>>(
        feat_con, g4, b4, wkv_h, bkv, Ykv);

    // 4) attention stage 1 (anchors over keys, online softmax)
    attn1_kernel<<<B_ * HEADS_, 128, 0, stream>>>(Ad, Ykv, out1T);

    // 5) attention stage 2 (queries over anchors)
    attn2_kernel<<<dim3(HW_/128, B_ * HEADS_), 256, 0, stream>>>(Yq, Ac, out1T, Yout2);

    // 6) output projection + transpose + residual
    final_gemm_kernel<<<dim3(32, 4, B_), 256, FG_SMEM, stream>>>(
        Yout2, wo_h, bo, feat_diff, out);
}
